// Attention_83193516523627
// MI455X (gfx1250) — matrix operand order repaired and checked by reading, not compiled, not run on hardware
//
#include <hip/hip_runtime.h>
#include <hip/hip_bf16.h>

// ---------------------------------------------------------------------------
// Types for CDNA5 WMMA (wave32): v_wmma_f32_16x16x32_bf16
// ---------------------------------------------------------------------------
typedef __attribute__((ext_vector_type(16))) __bf16 v16bf;
typedef __attribute__((ext_vector_type(8)))  __bf16 v8bf;
typedef __attribute__((ext_vector_type(8)))  float  v8f;

__device__ __forceinline__ v8f wmma_bf16(v16bf a, v16bf b, v8f c) {
  // 8 args: (neg_a, A, neg_b, B, c_mod, C, reuse_a, reuse_b)
  return __builtin_amdgcn_wmma_f32_16x16x32_bf16(
      false, a, false, b, (short)0, c, false, false);
}

// gfx1250 async copy: global memory -> LDS, 16 bytes per lane, ASYNCcnt.
// lds_addr = per-lane LDS byte address (low 32 bits of the flat LDS pointer),
// gptr     = per-lane 64-bit global address.
__device__ __forceinline__ void async_g2l_b128(unsigned lds_addr, const void* gptr) {
  asm volatile("global_load_async_to_lds_b128 %0, %1, off"
               :: "v"(lds_addr), "v"(gptr) : "memory");
}
__device__ __forceinline__ void wait_asynccnt0() {
  asm volatile("s_wait_asynccnt 0x0" ::: "memory");
}

#define DIM      768
#define HEADS    12
#define HEAD_D   64
#define BATCH    4
#define SEQ      2048
#define MROWS    (BATCH * SEQ)       // 8192
#define SCALE    0.125f              // 64^-0.5
#define KBLK     32                  // keys per flash iteration
#define NKB      (SEQ / KBLK)        // 64

// ---------------------------------------------------------------------------
// fp32 -> bf16 conversion (vectorized 4/thread)
// ---------------------------------------------------------------------------
__global__ void __launch_bounds__(256) cvt_f32_to_bf16(
    const float* __restrict__ s, __bf16* __restrict__ d, int n) {
  int i = (blockIdx.x * blockDim.x + threadIdx.x) * 4;
  if (i + 3 < n) {
    float4 f = *(const float4*)(s + i);
    d[i + 0] = (__bf16)f.x;
    d[i + 1] = (__bf16)f.y;
    d[i + 2] = (__bf16)f.z;
    d[i + 3] = (__bf16)f.w;
  } else {
    for (int j = i; j < n; ++j) d[j] = (__bf16)s[j];
  }
}

// ---------------------------------------------------------------------------
// QKV projection: out = x @ W^T, written per-head as [B,H,N,64] bf16.
// One wave per 16x16 output tile; 8 waves per 256-thread block.
// grid = (M/16, 768/128, 3) ; blockIdx.z selects {Wq,Wk,Wv}.
// ---------------------------------------------------------------------------
__global__ void __launch_bounds__(256) qkv_gemm(
    const __bf16* __restrict__ xb,
    const __bf16* __restrict__ wq, const __bf16* __restrict__ wk,
    const __bf16* __restrict__ wv,
    __bf16* __restrict__ q, __bf16* __restrict__ k, __bf16* __restrict__ v) {
  const __bf16* W;
  __bf16* O;
  if (blockIdx.z == 0)      { W = wq; O = q; }
  else if (blockIdx.z == 1) { W = wk; O = k; }
  else                      { W = wv; O = v; }

  const int mbase = blockIdx.x * 16;
  const int wave  = threadIdx.x >> 5;
  const int lane  = threadIdx.x & 31;
  const int h     = lane >> 4;          // K-half selector (ISA A/B layouts)
  const int ln    = lane & 15;
  const int colbase = blockIdx.y * 128 + wave * 16;

  const __bf16* arow = xb + (size_t)(mbase + ln) * DIM;
  const __bf16* brow = W  + (size_t)(colbase + ln) * DIM;

  v8f acc = {0.f, 0.f, 0.f, 0.f, 0.f, 0.f, 0.f, 0.f};

  for (int k0 = 0; k0 < DIM; k0 += 32) {
    // A 16x32 bf16: lane ln holds row ln; h=0 -> K {0..15}, h=1 -> {16..31}
    v8bf a0 = *(const v8bf*)(arow + k0 + 16 * h);
    v8bf a1 = *(const v8bf*)(arow + k0 + 8 + 16 * h);
    v16bf A;
    #pragma unroll
    for (int j = 0; j < 8; ++j) { A[j] = a0[j]; A[j + 8] = a1[j]; }
    // B 32x16 bf16: lane ln holds col ln; h selects K half; contiguous in W row
    v16bf B = *(const v16bf*)(brow + k0 + 16 * h);
    acc = wmma_bf16(A, B, acc);
    if (k0 + 64 < DIM) {
      __builtin_prefetch(arow + k0 + 64, 0, 3);
      __builtin_prefetch(brow + k0 + 64, 0, 3);
    }
  }

  // C/D layout: acc[r] -> row (r + 8h), col ln
  const int b = mbase / SEQ;
  const int col = colbase + ln;
  const int head = col >> 6, dd = col & 63;
  #pragma unroll
  for (int r = 0; r < 8; ++r) {
    int n = (mbase + r + 8 * h) & (SEQ - 1);
    size_t idx = (((size_t)(b * HEADS + head)) * SEQ + n) * HEAD_D + dd;
    O[idx] = (__bf16)acc[r];
  }
}

// ---------------------------------------------------------------------------
// Flash attention: softmax(Q K^T * scale) V  per (b,h).
// 128 threads = 4 waves; each wave owns a 16-query tile (WG covers 64 queries).
// grid = (SEQ/64, B*H). K and V blocks async-copied global->LDS (ASYNCcnt),
// V then transposed LDS->LDS for the P@V B-operand.
// Output: [B, N, 768] bf16 (head-concatenated).
// ---------------------------------------------------------------------------
__global__ void __launch_bounds__(128) flash_attn(
    const __bf16* __restrict__ Q, const __bf16* __restrict__ K,
    const __bf16* __restrict__ V, __bf16* __restrict__ AO) {
  __shared__ __align__(32) __bf16 sK[KBLK * HEAD_D];       // [key][dim]   (async dst)
  __shared__ __align__(32) __bf16 sV[KBLK * HEAD_D];       // [key][dim]   (async dst)
  __shared__ __align__(32) __bf16 sVT[HEAD_D * KBLK];      // [dim][key]
  __shared__ __align__(32) float  sS[4][16 * KBLK];        // per-wave scores
  __shared__ __align__(32) __bf16 sP[4][16 * KBLK];        // per-wave exp(P)
  __shared__ float sM[4][16], sL[4][16], sF[4][16];

  const int t    = threadIdx.x;
  const int wave = t >> 5;
  const int lane = t & 31;
  const int h    = lane >> 4;
  const int ln   = lane & 15;
  const int bh   = blockIdx.y;                 // 0..47
  const int b    = bh / HEADS, head = bh % HEADS;
  const int q0   = blockIdx.x * 64 + wave * 16;
  const size_t base = (size_t)bh * SEQ * HEAD_D;

  // Per-lane LDS byte addresses for the async destinations.
  const unsigned sK_lds = (unsigned)(uintptr_t)&sK[0];
  const unsigned sV_lds = (unsigned)(uintptr_t)&sV[0];

  if (t < 64) { sM[t >> 4][t & 15] = -__builtin_inff(); sL[t >> 4][t & 15] = 0.f; }

  // Load this wave's Q tile (16x64) as two 16x32 A-operands, kept in VGPRs.
  v16bf aq[2];
  {
    const __bf16* qrow = Q + base + (size_t)(q0 + ln) * HEAD_D;
    #pragma unroll
    for (int kc = 0; kc < 2; ++kc) {
      v8bf lo = *(const v8bf*)(qrow + kc * 32 + 16 * h);
      v8bf hi = *(const v8bf*)(qrow + kc * 32 + 8 + 16 * h);
      #pragma unroll
      for (int j = 0; j < 8; ++j) { aq[kc][j] = lo[j]; aq[kc][j + 8] = hi[j]; }
    }
  }

  v8f Oacc[4];
  #pragma unroll
  for (int c = 0; c < 4; ++c)
    #pragma unroll
    for (int r = 0; r < 8; ++r) Oacc[c][r] = 0.f;

  for (int kb = 0; kb < NKB; ++kb) {
    __syncthreads();  // protect sK/sV/sVT reuse across iterations
    {
      // Async stage: K block (4KB) and V block (4KB), both contiguous in
      // [B,H,N,64]. 128 threads x 16B x 2 rounds each = 4KB per tensor.
      const char* kg = (const char*)(K + base + (size_t)kb * KBLK * HEAD_D);
      const char* vg = (const char*)(V + base + (size_t)kb * KBLK * HEAD_D);
      #pragma unroll
      for (int j = 0; j < 2; ++j) {
        unsigned boff = (unsigned)(j * 128 + t) * 16;
        async_g2l_b128(sK_lds + boff, kg + boff);
        async_g2l_b128(sV_lds + boff, vg + boff);
      }
      wait_asynccnt0();
    }
    __syncthreads();
    {
      // Transpose V in LDS: sVT[dim][key] = sV[key][dim]
      int row = t >> 2, c0 = (t & 3) * 16;
      v16bf vv = *(const v16bf*)&sV[row * HEAD_D + c0];
      #pragma unroll
      for (int j = 0; j < 16; ++j) sVT[(c0 + j) * KBLK + row] = vv[j];
    }
    __syncthreads();

    // S (16x32) = Q (16x64) @ K_blk^T : 4 WMMAs
    v8f s0 = {0.f,0.f,0.f,0.f,0.f,0.f,0.f,0.f};
    v8f s1 = {0.f,0.f,0.f,0.f,0.f,0.f,0.f,0.f};
    #pragma unroll
    for (int kc = 0; kc < 2; ++kc) {
      v16bf b0 = *(const v16bf*)&sK[ln * HEAD_D + kc * 32 + 16 * h];
      s0 = wmma_bf16(aq[kc], b0, s0);
      v16bf b1 = *(const v16bf*)&sK[(16 + ln) * HEAD_D + kc * 32 + 16 * h];
      s1 = wmma_bf16(aq[kc], b1, s1);
    }
    // C/D layout -> LDS row-major for softmax / transpose to A layout
    #pragma unroll
    for (int r = 0; r < 8; ++r) {
      sS[wave][(r + 8 * h) * KBLK + ln]      = s0[r];
      sS[wave][(r + 8 * h) * KBLK + 16 + ln] = s1[r];
    }
    __syncthreads();

    // Online softmax: lanes 0..15 each own one query row.
    if (lane < 16) {
      const float* Sr = &sS[wave][ln * KBLK];
      float mx = -__builtin_inff();
      #pragma unroll
      for (int j = 0; j < KBLK; ++j) mx = fmaxf(mx, Sr[j] * SCALE);
      float mold = sM[wave][ln];
      float mnew = fmaxf(mold, mx);
      float f    = __expf(mold - mnew);
      float sum  = 0.f;
      #pragma unroll
      for (int j = 0; j < KBLK; ++j) {
        float p = __expf(Sr[j] * SCALE - mnew);
        sum += p;
        sP[wave][ln * KBLK + j] = (__bf16)p;
      }
      sM[wave][ln] = mnew;
      sL[wave][ln] = sL[wave][ln] * f + sum;
      sF[wave][ln] = f;
    }
    __syncthreads();

    // Rescale O accumulators by per-row factor.
    float fr[8];
    #pragma unroll
    for (int r = 0; r < 8; ++r) fr[r] = sF[wave][r + 8 * h];
    #pragma unroll
    for (int c = 0; c < 4; ++c)
      #pragma unroll
      for (int r = 0; r < 8; ++r) Oacc[c][r] *= fr[r];

    // O += P (16x32) @ V_blk (32x64) : 4 WMMAs (A from sP, B from sVT rows)
    v16bf ap;
    {
      v8bf lo = *(const v8bf*)&sP[wave][ln * KBLK + 16 * h];
      v8bf hi = *(const v8bf*)&sP[wave][ln * KBLK + 8 + 16 * h];
      #pragma unroll
      for (int j = 0; j < 8; ++j) { ap[j] = lo[j]; ap[j + 8] = hi[j]; }
    }
    #pragma unroll
    for (int c = 0; c < 4; ++c) {
      v16bf bv = *(const v16bf*)&sVT[(c * 16 + ln) * KBLK + 16 * h];
      Oacc[c] = wmma_bf16(ap, bv, Oacc[c]);
    }
  }
  __syncthreads();

  // Normalize and store to [B, N, H*64] bf16.
  float linv[8];
  #pragma unroll
  for (int r = 0; r < 8; ++r) linv[r] = 1.0f / sL[wave][r + 8 * h];
  #pragma unroll
  for (int c = 0; c < 4; ++c)
    #pragma unroll
    for (int r = 0; r < 8; ++r) {
      int row = q0 + r + 8 * h;
      int col = head * HEAD_D + c * 16 + ln;
      AO[((size_t)(b * SEQ + row)) * DIM + col] = (__bf16)(Oacc[c][r] * linv[r]);
    }
}

// ---------------------------------------------------------------------------
// Output projection: d_out = AO @ Wp^T + bp   (f32 out)
// ---------------------------------------------------------------------------
__global__ void __launch_bounds__(256) proj_gemm(
    const __bf16* __restrict__ ab, const __bf16* __restrict__ wp,
    const float* __restrict__ bias, float* __restrict__ out) {
  const int mbase = blockIdx.x * 16;
  const int wave  = threadIdx.x >> 5;
  const int lane  = threadIdx.x & 31;
  const int h     = lane >> 4;
  const int ln    = lane & 15;
  const int colbase = blockIdx.y * 128 + wave * 16;

  const __bf16* arow = ab + (size_t)(mbase + ln) * DIM;
  const __bf16* brow = wp + (size_t)(colbase + ln) * DIM;

  v8f acc = {0.f, 0.f, 0.f, 0.f, 0.f, 0.f, 0.f, 0.f};
  for (int k0 = 0; k0 < DIM; k0 += 32) {
    v8bf a0 = *(const v8bf*)(arow + k0 + 16 * h);
    v8bf a1 = *(const v8bf*)(arow + k0 + 8 + 16 * h);
    v16bf A;
    #pragma unroll
    for (int j = 0; j < 8; ++j) { A[j] = a0[j]; A[j + 8] = a1[j]; }
    v16bf B = *(const v16bf*)(brow + k0 + 16 * h);
    acc = wmma_bf16(A, B, acc);
    if (k0 + 64 < DIM) {
      __builtin_prefetch(arow + k0 + 64, 0, 3);
      __builtin_prefetch(brow + k0 + 64, 0, 3);
    }
  }

  const int col = colbase + ln;
  const float bv = bias[col];
  #pragma unroll
  for (int r = 0; r < 8; ++r) {
    int row = mbase + r + 8 * h;
    out[(size_t)row * DIM + col] = acc[r] + bv;
  }
}

// ---------------------------------------------------------------------------
// Host launcher
// ---------------------------------------------------------------------------
extern "C" void kernel_launch(void* const* d_in, const int* in_sizes, int n_in,
                              void* d_out, int out_size, void* d_ws, size_t ws_size,
                              hipStream_t stream) {
  const float* x  = (const float*)d_in[0];
  const float* Wq = (const float*)d_in[1];
  const float* Wk = (const float*)d_in[2];
  const float* Wv = (const float*)d_in[3];
  const float* Wp = (const float*)d_in[4];
  const float* bp = (const float*)d_in[5];
  float* out = (float*)d_out;

  char* ws = (char*)d_ws;
  size_t off = 0;
  auto nxt = [&](size_t bytes) -> char* {
    char* p = ws + off;
    off += (bytes + 255) & ~(size_t)255;
    return p;
  };
  const size_t XE = (size_t)MROWS * DIM;      // 6291456 elems
  const size_t WE = (size_t)DIM * DIM;        // 589824 elems
  const size_t QE = (size_t)BATCH * HEADS * SEQ * HEAD_D;  // == XE

  __bf16* xb  = (__bf16*)nxt(XE * 2);
  __bf16* wqb = (__bf16*)nxt(WE * 2);
  __bf16* wkb = (__bf16*)nxt(WE * 2);
  __bf16* wvb = (__bf16*)nxt(WE * 2);
  __bf16* wpb = (__bf16*)nxt(WE * 2);
  __bf16* qb  = (__bf16*)nxt(QE * 2);
  __bf16* kb  = (__bf16*)nxt(QE * 2);
  __bf16* vb  = (__bf16*)nxt(QE * 2);
  __bf16* aob = (__bf16*)nxt(XE * 2);

  // 1) fp32 -> bf16 conversions
  {
    int nx = (int)XE, nw = (int)WE;
    int bx = (nx / 4 + 255) / 256, bw = (nw / 4 + 255) / 256;
    cvt_f32_to_bf16<<<bx, 256, 0, stream>>>(x,  xb,  nx);
    cvt_f32_to_bf16<<<bw, 256, 0, stream>>>(Wq, wqb, nw);
    cvt_f32_to_bf16<<<bw, 256, 0, stream>>>(Wk, wkb, nw);
    cvt_f32_to_bf16<<<bw, 256, 0, stream>>>(Wv, wvb, nw);
    cvt_f32_to_bf16<<<bw, 256, 0, stream>>>(Wp, wpb, nw);
  }

  // 2) Q/K/V projections (WMMA GEMM), per-head layout
  {
    dim3 grid(MROWS / 16, DIM / 128, 3);
    qkv_gemm<<<grid, 256, 0, stream>>>(xb, wqb, wkb, wvb, qb, kb, vb);
  }

  // 3) Flash attention
  {
    dim3 grid(SEQ / 64, BATCH * HEADS);
    flash_attn<<<grid, 128, 0, stream>>>(qb, kb, vb, aob);
  }

  // 4) Output projection + bias (f32 out)
  {
    dim3 grid(MROWS / 16, DIM / 128);
    proj_gemm<<<grid, 256, 0, stream>>>(aob, wpb, bp, out);
  }
}
